// LoRAMultiheadAttention1_61581241090659
// MI455X (gfx1250) — matrix operand order repaired and checked by reading, not compiled, not run on hardware
//
#include <hip/hip_runtime.h>
#include <hip/hip_bf16.h>

// ---------------------------------------------------------------------------
// LoRA multi-head attention (SAM-style decomposed rel-pos), MI455X / gfx1250.
//
// Pipeline (all on `stream`, scratch in d_ws):
//   1. convert x, W_qkv, W_proj fp32 -> bf16
//   2. tiny LoRA stage-1: xa_q = x@A_q^T, xa_v = x@A_v^T   (4096x4 each)
//   3. QKV GEMM via v_wmma_f32_16x16x32_bf16, LoRA+bias fused in epilogue,
//      scatter to q(head,n,64), k(head,n,64), v^T(head,64,n)  [bf16]
//   4. rel-pos bias tables bias_h/bias_w (head, token, 64)    [fp32]
//   5. flash attention: per-wave 16-query tile, stream 32-key blocks,
//      S and O tiles through WMMA, online softmax via shfl_xor reductions,
//      P transposed C-layout -> A-layout through a 1KB LDS tile per wave.
//   6. out-proj GEMM via WMMA, fp32 output.
// ---------------------------------------------------------------------------

typedef __bf16 bf16;
typedef __attribute__((ext_vector_type(16))) __bf16 v16bf;
typedef __attribute__((ext_vector_type(8)))  __bf16 v8bf;
typedef __attribute__((ext_vector_type(8)))  float  v8f;
typedef __attribute__((ext_vector_type(4)))  float  v4f;

#define DIM    768
#define NHEAD  12
#define HD     64
#define NTOK   4096     // 64*64 tokens
#define QKV_N  (3*DIM)

// ---------------------------------------------------------------------------
// fp32 -> bf16 elementwise convert
// ---------------------------------------------------------------------------
__global__ void cvt_f32_bf16_kernel(const float* __restrict__ in,
                                    bf16* __restrict__ out, int n) {
  int i = blockIdx.x * blockDim.x + threadIdx.x;
  if (i < n) out[i] = (bf16)in[i];
}

// ---------------------------------------------------------------------------
// LoRA stage 1: xa_q[m][r] = x[m]·A_q[r],  xa_v[m][r] = x[m]·A_v[r]  (r<4)
// ---------------------------------------------------------------------------
__global__ void lora_xa_kernel(const float* __restrict__ x,
                               const float* __restrict__ Aq,
                               const float* __restrict__ Av,
                               float* __restrict__ xaq,
                               float* __restrict__ xav) {
  int t = blockIdx.x * blockDim.x + threadIdx.x;
  if (t >= NTOK * 8) return;
  int m = t >> 3, j = t & 7;
  const float* a  = (j < 4) ? (Aq + (j & 3) * DIM) : (Av + (j & 3) * DIM);
  const float* xr = x + (size_t)m * DIM;
  float s = 0.f;
  for (int k = 0; k < DIM; k += 4) {
    v4f xv = *(const v4f*)(xr + k);
    v4f av = *(const v4f*)(a + k);
    s += xv.x * av.x + xv.y * av.y + xv.z * av.z + xv.w * av.w;
  }
  if (j < 4) xaq[m * 4 + j] = s; else xav[m * 4 + (j - 4)] = s;
}

// ---------------------------------------------------------------------------
// A-fragment loader: lane<16 holds row lane, K = {0..15};
// lane>=16 holds row lane-16, K = {16..31}. (contiguous K order, matches B)
// ---------------------------------------------------------------------------
__device__ __forceinline__ v16bf load_a_frag(const bf16* row, int lane) {
  int b0 = (lane < 16) ? 0 : 16;
  v8bf lo = *(const v8bf*)(row + b0);
  v8bf hi = *(const v8bf*)(row + b0 + 8);
  v16bf a;
#pragma unroll
  for (int i = 0; i < 8; ++i) { a[i] = lo[i]; a[i + 8] = hi[i]; }
  return a;
}

// ---------------------------------------------------------------------------
// QKV GEMM + fused LoRA/bias epilogue.
// C[m,f] = x[m]·W_qkv[f] + b[f] (+ lora), scattered to q / k / v^T (bf16).
// Wave computes a 16x64 tile; block of 4 waves covers 64x64.
// ---------------------------------------------------------------------------
__global__ void __launch_bounds__(128)
qkv_gemm_kernel(const bf16* __restrict__ xbf, const bf16* __restrict__ wbf,
                const float* __restrict__ bqkv,
                const float* __restrict__ xaq, const float* __restrict__ xav,
                const float* __restrict__ Bq, const float* __restrict__ Bv,
                bf16* __restrict__ qb, bf16* __restrict__ kb,
                bf16* __restrict__ vtb) {
  const int lane = threadIdx.x & 31, wave = threadIdx.x >> 5;
  const int m0 = blockIdx.x * 64 + wave * 16;
  const int n0 = blockIdx.y * 64;
  const int hl = lane >> 4, ll = lane & 15;

  v8f acc[4] = {};
  const bf16* arow = xbf + (size_t)(m0 + ll) * DIM;
  for (int k0 = 0; k0 < DIM; k0 += 32) {
    v16bf a = load_a_frag(arow + k0, lane);
#pragma unroll
    for (int g = 0; g < 4; ++g) {
      // B operand: lane = output column f; 16 contiguous K along W row.
      const bf16* bp = wbf + (size_t)(n0 + g * 16 + ll) * DIM + k0 + hl * 16;
      v16bf b = *(const v16bf*)bp;
      acc[g] = __builtin_amdgcn_wmma_f32_16x16x32_bf16(
          false, a, false, b, (short)0, acc[g], false, false);
    }
  }
#pragma unroll
  for (int g = 0; g < 4; ++g) {
    int f = n0 + g * 16 + ll;
#pragma unroll
    for (int r = 0; r < 8; ++r) {
      int m = m0 + r + 8 * hl;
      float v = acc[g][r] + bqkv[f];
      if (f < DIM) {                       // q third: + (xa_q @ B_q^T)/rank
        v4f xa = *(const v4f*)(xaq + m * 4);
        v4f bb = *(const v4f*)(Bq + f * 4);
        v += 0.25f * (xa.x * bb.x + xa.y * bb.y + xa.z * bb.z + xa.w * bb.w);
      } else if (f >= 2 * DIM) {           // v third: + (xa_v @ B_v^T)/rank
        v4f xa = *(const v4f*)(xav + m * 4);
        v4f bb = *(const v4f*)(Bv + (f - 2 * DIM) * 4);
        v += 0.25f * (xa.x * bb.x + xa.y * bb.y + xa.z * bb.z + xa.w * bb.w);
      }
      int part = f / DIM;
      int fin = f - part * DIM;
      int head = fin >> 6, c = fin & 63;
      if (part == 0)      qb[((size_t)head * NTOK + m) * HD + c] = (bf16)v;
      else if (part == 1) kb[((size_t)head * NTOK + m) * HD + c] = (bf16)v;
      else                vtb[((size_t)head * HD + c) * NTOK + m] = (bf16)v;  // V^T
    }
  }
}

// ---------------------------------------------------------------------------
// Decomposed rel-pos bias: bias[head][tok][k] = q[head,tok]·rel[qc - k + 63]
// use_h: qc = qh (row coord), else qc = qw (col coord).
// ---------------------------------------------------------------------------
__global__ void __launch_bounds__(64)
relbias_kernel(const bf16* __restrict__ qb, const float* __restrict__ rel,
               float* __restrict__ bias, int use_h) {
  __shared__ float qs[HD];
  int head = blockIdx.x >> 12;
  int tok  = blockIdx.x & 4095;
  int qh = tok >> 6, qw = tok & 63;
  int t = threadIdx.x;                  // key coord 0..63
  qs[t] = (float)qb[((size_t)head * NTOK + tok) * HD + t];
  __syncthreads();
  int qc = use_h ? qh : qw;
  const float* rr = rel + (size_t)(qc - t + 63) * HD;
  float s = 0.f;
#pragma unroll
  for (int c = 0; c < HD; c += 4)
    s += qs[c] * rr[c] + qs[c+1] * rr[c+1] + qs[c+2] * rr[c+2] + qs[c+3] * rr[c+3];
  bias[((size_t)head * NTOK + tok) * HD + t] = s;
}

// ---------------------------------------------------------------------------
// Flash attention. grid = (head, qtile64); block = 128 (4 waves).
// Each wave owns 16 queries, streams 4096 keys in 32-key blocks.
// ---------------------------------------------------------------------------
__global__ void __launch_bounds__(128)
flash_attn_kernel(const bf16* __restrict__ qb, const bf16* __restrict__ kb,
                  const bf16* __restrict__ vtb, const float* __restrict__ bh,
                  const float* __restrict__ bw, bf16* __restrict__ ob) {
  __shared__ float bh_s[4][16 * HD];   // per-wave bias_h tile (16 q x 64 kh)
  __shared__ float bw_s[4][16 * HD];   // per-wave bias_w tile (16 q x 64 kw)
  __shared__ bf16  p_s[4][16 * 32];    // per-wave P tile for C->A relayout

  const int lane = threadIdx.x & 31, wave = threadIdx.x >> 5;
  const int head = blockIdx.x;
  const int qbase = blockIdx.y * 64 + wave * 16;
  const int hl = lane >> 4, ll = lane & 15;
  const float scale = 0.125f;          // hd^-0.5 = 1/8

  for (int i = lane; i < 16 * HD; i += 32) {
    int q = qbase + (i >> 6);
    bh_s[wave][i] = bh[((size_t)head * NTOK + q) * HD + (i & 63)];
    bw_s[wave][i] = bw[((size_t)head * NTOK + q) * HD + (i & 63)];
  }
  __syncthreads();

  const bf16* qrow = qb + ((size_t)head * NTOK + qbase + ll) * HD;
  v16bf qa0 = load_a_frag(qrow, lane);        // c 0..31
  v16bf qa1 = load_a_frag(qrow + 32, lane);   // c 32..63

  v8f o[4] = {};
  float mrow[8], lrow[8];
#pragma unroll
  for (int r = 0; r < 8; ++r) { mrow[r] = -3.0e38f; lrow[r] = 0.f; }

  float* bhw = bh_s[wave];
  float* bww = bw_s[wave];
  bf16*  pw  = p_s[wave];

  for (int blk = 0; blk < NTOK; blk += 32) {
    // K operand: lane = key column; contiguous along c within a key row.
    const bf16* k0p = kb + ((size_t)head * NTOK + blk + ll) * HD + hl * 16;
    const bf16* k1p = k0p + 16 * HD;
    __builtin_prefetch(k0p + 32 * HD, 0, 3);       // next 32-key block
    v16bf kb00 = *(const v16bf*)(k0p);
    v16bf kb01 = *(const v16bf*)(k0p + 32);
    v16bf kb10 = *(const v16bf*)(k1p);
    v16bf kb11 = *(const v16bf*)(k1p + 32);

    v8f s0 = {}, s1 = {};
    s0 = __builtin_amdgcn_wmma_f32_16x16x32_bf16(false, qa0, false, kb00, (short)0, s0, false, false);
    s0 = __builtin_amdgcn_wmma_f32_16x16x32_bf16(false, qa1, false, kb01, (short)0, s0, false, false);
    s1 = __builtin_amdgcn_wmma_f32_16x16x32_bf16(false, qa0, false, kb10, (short)0, s1, false, false);
    s1 = __builtin_amdgcn_wmma_f32_16x16x32_bf16(false, qa1, false, kb11, (short)0, s1, false, false);

    const int kh  = blk >> 6;   // 32 keys never straddle a 64-wide image row
    const int kwb = blk & 63;

#pragma unroll
    for (int r = 0; r < 8; ++r) {
      int row = r + 8 * hl;
      float bcom = bhw[row * 64 + kh];
      float v0 = s0[r] * scale + bcom + bww[row * 64 + kwb + ll];
      float v1 = s1[r] * scale + bcom + bww[row * 64 + kwb + 16 + ll];
      // row max across the 16 lanes holding this row (xor<=8 stays in half)
      float t = fmaxf(v0, v1);
      t = fmaxf(t, __shfl_xor(t, 1));
      t = fmaxf(t, __shfl_xor(t, 2));
      t = fmaxf(t, __shfl_xor(t, 4));
      t = fmaxf(t, __shfl_xor(t, 8));
      float mn = fmaxf(mrow[r], t);
      float alpha = __expf(mrow[r] - mn);
      mrow[r] = mn;
      float p0 = __expf(v0 - mn), p1 = __expf(v1 - mn);
      float su = p0 + p1;
      su += __shfl_xor(su, 1);
      su += __shfl_xor(su, 2);
      su += __shfl_xor(su, 4);
      su += __shfl_xor(su, 8);
      lrow[r] = lrow[r] * alpha + su;
#pragma unroll
      for (int g = 0; g < 4; ++g) o[g][r] *= alpha;
      pw[row * 32 + ll]      = (bf16)p0;
      pw[row * 32 + 16 + ll] = (bf16)p1;
    }
    asm volatile("s_wait_dscnt 0" ::: "memory");   // LDS WAR/RAW within wave

    // P as A operand (16 q x 32 keys) from LDS
    v16bf pa;
    {
      const bf16* prow = pw + ll * 32;
      int b0 = hl ? 16 : 0;
      v8bf lo = *(const v8bf*)(prow + b0);
      v8bf hi = *(const v8bf*)(prow + b0 + 8);
#pragma unroll
      for (int i = 0; i < 8; ++i) { pa[i] = lo[i]; pa[i + 8] = hi[i]; }
    }
    // V^T as B operand: lane = output channel c; contiguous along keys.
#pragma unroll
    for (int g = 0; g < 4; ++g) {
      const bf16* vp = vtb + ((size_t)head * HD + g * 16 + ll) * NTOK + blk + hl * 16;
      __builtin_prefetch(vp + 32, 0, 3);
      v16bf vb = *(const v16bf*)vp;
      o[g] = __builtin_amdgcn_wmma_f32_16x16x32_bf16(
          false, pa, false, vb, (short)0, o[g], false, false);
    }
  }

#pragma unroll
  for (int r = 0; r < 8; ++r) {
    float inv = 1.f / lrow[r];
    int tok = qbase + r + 8 * hl;
#pragma unroll
    for (int g = 0; g < 4; ++g)
      ob[(size_t)tok * DIM + head * HD + g * 16 + ll] = (bf16)(o[g][r] * inv);
  }
}

// ---------------------------------------------------------------------------
// Output projection GEMM: out[m,f] = attn[m]·W_proj[f] + b_proj[f]  (fp32 out)
// ---------------------------------------------------------------------------
__global__ void __launch_bounds__(128)
proj_gemm_kernel(const bf16* __restrict__ abf, const bf16* __restrict__ wbf,
                 const float* __restrict__ bias, float* __restrict__ out) {
  const int lane = threadIdx.x & 31, wave = threadIdx.x >> 5;
  const int m0 = blockIdx.x * 64 + wave * 16;
  const int n0 = blockIdx.y * 64;
  const int hl = lane >> 4, ll = lane & 15;

  v8f acc[4] = {};
  const bf16* arow = abf + (size_t)(m0 + ll) * DIM;
  for (int k0 = 0; k0 < DIM; k0 += 32) {
    v16bf a = load_a_frag(arow + k0, lane);
#pragma unroll
    for (int g = 0; g < 4; ++g) {
      const bf16* bp = wbf + (size_t)(n0 + g * 16 + ll) * DIM + k0 + hl * 16;
      v16bf b = *(const v16bf*)bp;
      acc[g] = __builtin_amdgcn_wmma_f32_16x16x32_bf16(
          false, a, false, b, (short)0, acc[g], false, false);
    }
  }
#pragma unroll
  for (int g = 0; g < 4; ++g) {
    int f = n0 + g * 16 + ll;
    float bv = bias[f];
#pragma unroll
    for (int r = 0; r < 8; ++r) {
      int m = m0 + r + 8 * hl;
      out[(size_t)m * DIM + f] = acc[g][r] + bv;
    }
  }
}

// ---------------------------------------------------------------------------
extern "C" void kernel_launch(void* const* d_in, const int* in_sizes, int n_in,
                              void* d_out, int out_size, void* d_ws, size_t ws_size,
                              hipStream_t stream) {
  const float* x     = (const float*)d_in[0];
  const float* Wqkv  = (const float*)d_in[1];
  const float* bqkv  = (const float*)d_in[2];
  const float* Aq    = (const float*)d_in[3];
  const float* Bq    = (const float*)d_in[4];
  const float* Av    = (const float*)d_in[5];
  const float* Bv    = (const float*)d_in[6];
  const float* relh  = (const float*)d_in[7];
  const float* relw  = (const float*)d_in[8];
  const float* Wproj = (const float*)d_in[9];
  const float* bproj = (const float*)d_in[10];
  float* out = (float*)d_out;

  char* ws = (char*)d_ws;
  size_t off = 0;
  auto alloc = [&](size_t bytes) -> char* {
    char* p = ws + off;
    off += (bytes + 255) & ~(size_t)255;
    return p;
  };
  bf16*  xbf     = (bf16*)alloc((size_t)NTOK * DIM * 2);
  bf16*  wqkvbf  = (bf16*)alloc((size_t)QKV_N * DIM * 2);
  bf16*  wprojbf = (bf16*)alloc((size_t)DIM * DIM * 2);
  float* xaq     = (float*)alloc((size_t)NTOK * 4 * 4);
  float* xav     = (float*)alloc((size_t)NTOK * 4 * 4);
  bf16*  qbuf    = (bf16*)alloc((size_t)NHEAD * NTOK * HD * 2);
  bf16*  kbuf    = (bf16*)alloc((size_t)NHEAD * NTOK * HD * 2);
  bf16*  vtbuf   = (bf16*)alloc((size_t)NHEAD * HD * NTOK * 2);
  float* biash   = (float*)alloc((size_t)NHEAD * NTOK * HD * 4);
  float* biasw   = (float*)alloc((size_t)NHEAD * NTOK * HD * 4);
  bf16*  aobf    = (bf16*)alloc((size_t)NTOK * DIM * 2);

  int n1 = NTOK * DIM;
  cvt_f32_bf16_kernel<<<(n1 + 255) / 256, 256, 0, stream>>>(x, xbf, n1);
  int n2 = QKV_N * DIM;
  cvt_f32_bf16_kernel<<<(n2 + 255) / 256, 256, 0, stream>>>(Wqkv, wqkvbf, n2);
  int n3 = DIM * DIM;
  cvt_f32_bf16_kernel<<<(n3 + 255) / 256, 256, 0, stream>>>(Wproj, wprojbf, n3);

  lora_xa_kernel<<<(NTOK * 8 + 255) / 256, 256, 0, stream>>>(x, Aq, Av, xaq, xav);

  qkv_gemm_kernel<<<dim3(NTOK / 64, QKV_N / 64), 128, 0, stream>>>(
      xbf, wqkvbf, bqkv, xaq, xav, Bq, Bv, qbuf, kbuf, vtbuf);

  relbias_kernel<<<NHEAD * NTOK, 64, 0, stream>>>(qbuf, relh, biash, 1);
  relbias_kernel<<<NHEAD * NTOK, 64, 0, stream>>>(qbuf, relw, biasw, 0);

  flash_attn_kernel<<<dim3(NHEAD, NTOK / 64), 128, 0, stream>>>(
      qbuf, kbuf, vtbuf, biash, biasw, aobf);

  proj_gemm_kernel<<<dim3(NTOK / 64, DIM / 64), 128, 0, stream>>>(
      aobf, wprojbf, bproj, out);
}
